// MultiHeadAttention_15272903705202
// MI455X (gfx1250) — compile-verified, matrix operand order checked by reading, not run on hardware
//
#include <hip/hip_runtime.h>
#include <hip/hip_bf16.h>
#include <stdint.h>

// MHA forward for MI455X (gfx1250), wave32 + WMMA bf16.
// Stage 1: fused QKV projection GEMM (fp32 in -> bf16 Q/K/V in workspace),
//          double-buffered LDS tiles so global loads overlap WMMA.
// Stage 2: flash attention, one 16-row q tile per wave, 128-wide kv chunks,
//          online softmax, mask tiles double-buffered in LDS via
//          global_load_async_to_lds_b128 (ASYNCcnt), P transposed through
//          LDS, all matmuls via v_wmma_f32_16x16x32_bf16.
// bf16 conversion: RNE bias add (v_bfe+v_add3) + v_perm_b32 byte-merge.

typedef __attribute__((ext_vector_type(16))) __bf16        v16bf;
typedef __attribute__((ext_vector_type(8)))  float         v8f;
typedef __attribute__((ext_vector_type(8)))  unsigned int  v8u;

#define NEGV (-1.0e9f)

// fp32 -> bf16 with round-to-nearest-even; result lives in bits [31:16].
__device__ __forceinline__ unsigned int f2bf_hi(float f) {
  unsigned int u = __float_as_uint(f);
  return u + (0x7FFFu + ((u >> 16) & 1u));
}
// One 16-bit bf16 value (for scalar stores).
__device__ __forceinline__ unsigned short f2bf(float f) {
  return (unsigned short)(f2bf_hi(f) >> 16);
}
// Two fp32 -> packed 2xbf16 in one dword: v_perm_b32 merges the high halves.
__device__ __forceinline__ unsigned int pack2_bf16(float lo, float hi) {
  return __builtin_amdgcn_perm(f2bf_hi(hi), f2bf_hi(lo), 0x07060302u);
}

__device__ __forceinline__ v8f bf_wmma(v16bf a, v16bf b, v8f c) {
  // (neg_a, A, neg_b, B, c_mod, C, reuse_a, reuse_b)
  return __builtin_amdgcn_wmma_f32_16x16x32_bf16(false, a, false, b, (short)0, c, false, false);
}

// Load a 16x32 bf16 A/B fragment from a buffer that is row-major along the
// reduction (k) dimension. ISA layout: lane l -> row (l&15); VGPR0..3 hold
// k = kb..kb+7, VGPR4..7 hold k = kb+16..kb+23, kb = (l>>4)*8.
// => two 16-byte loads per lane. Works for LDS and global pointers.
__device__ __forceinline__ v16bf load_frag(const unsigned short* p, int stride) {
  const int lane = threadIdx.x & 31;
  const int row  = lane & 15;
  const int kb   = (lane >> 4) << 3;
  const unsigned short* q = p + (size_t)row * stride + kb;
  uint4 lo = *(const uint4*)(q);
  uint4 hi = *(const uint4*)(q + 16);
  v8u u;
  u[0] = lo.x; u[1] = lo.y; u[2] = lo.z; u[3] = lo.w;
  u[4] = hi.x; u[5] = hi.y; u[6] = hi.z; u[7] = hi.w;
  return __builtin_bit_cast(v16bf, u);
}

// ---------------------------------------------------------------------------
// Stage 1: Y = X*W + bias, fp32 inputs, bf16 outputs.
// Block tile 128(M) x 64(N), BK=32, double-buffered LDS. 256 threads = 8
// waves, each wave a 32x32 sub-tile. blockIdx.z: 0=Q, 1=K, 2=V (V transposed).
// ---------------------------------------------------------------------------
__global__ __launch_bounds__(256) void qkv_proj_kernel(
    const float* __restrict__ x_q, const float* __restrict__ x_kv,
    const float* __restrict__ w_q, const float* __restrict__ w_k,
    const float* __restrict__ w_v,
    const float* __restrict__ b_q, const float* __restrict__ b_k,
    const float* __restrict__ b_v,
    unsigned short* __restrict__ qws, unsigned short* __restrict__ kws,
    unsigned short* __restrict__ vtws)
{
  const int proj = blockIdx.z;
  const float* X    = (proj == 0) ? x_q : x_kv;
  const float* W    = (proj == 0) ? w_q : (proj == 1) ? w_k : w_v;
  const float* bias = (proj == 0) ? b_q : (proj == 1) ? b_k : b_v;
  unsigned short* out = (proj == 0) ? qws : (proj == 1) ? kws : vtws;

  __shared__ unsigned short As[2][128 * 32];   // [m][k]
  __shared__ unsigned short Bs[2][64 * 32];    // [n][k]  (W^T tile)

  const int tid   = threadIdx.x;
  const int wave  = tid >> 5;
  const int lane  = tid & 31;
  const int mbase = blockIdx.y * 128;
  const int nbase = blockIdx.x * 64;
  const int wm0   = (wave & 3) * 32;
  const int wn0   = (wave >> 2) * 32;

  // A-tile per-thread coordinates (constant-folded).
  int rowA[4], c4A[4];
  #pragma unroll
  for (int j = 0; j < 4; ++j) {
    int idx = tid + j * 256;
    rowA[j] = idx >> 3;
    c4A[j]  = (idx & 7) << 2;
  }
  // B-tile: thread owns one n column and two k-quads {kq0, kq0+16}.
  const int nB  = tid & 63;
  const int kq0 = (tid >> 6) << 2;            // 0,4,8,12

  float4 fa[4];
  float  fbv[2][4];

  // Issue all global loads for a k-slab into registers (no waits in between),
  // plus unconditional speculative prefetch of the following slab.
  auto load_tile = [&](int kk) {
    #pragma unroll
    for (int j = 0; j < 4; ++j) {
      const float* p = X + (size_t)(mbase + rowA[j]) * 1024 + kk + c4A[j];
      fa[j] = *(const float4*)p;
      __builtin_prefetch(p + 32, 0, 1);   // next k-slab, speculative
    }
    #pragma unroll
    for (int j = 0; j < 2; ++j) {
      const int kq = kq0 + j * 16;
      const float* p = W + (size_t)(kk + kq) * 1024 + nbase + nB;
      #pragma unroll
      for (int e = 0; e < 4; ++e) fbv[j][e] = p[(size_t)e * 1024];
      __builtin_prefetch(p + (size_t)32 * 1024, 0, 1);
    }
  };

  auto store_tile = [&](int buf) {
    #pragma unroll
    for (int j = 0; j < 4; ++j) {
      unsigned int u0 = pack2_bf16(fa[j].x, fa[j].y);
      unsigned int u1 = pack2_bf16(fa[j].z, fa[j].w);
      *(uint2*)(&As[buf][rowA[j] * 32 + c4A[j]]) = make_uint2(u0, u1);
    }
    #pragma unroll
    for (int j = 0; j < 2; ++j) {
      unsigned int u0 = pack2_bf16(fbv[j][0], fbv[j][1]);
      unsigned int u1 = pack2_bf16(fbv[j][2], fbv[j][3]);
      *(uint2*)(&Bs[buf][nB * 32 + kq0 + j * 16]) = make_uint2(u0, u1);
    }
  };

  const v8f zc = {};
  v8f acc[2][2];
  acc[0][0] = zc; acc[0][1] = zc; acc[1][0] = zc; acc[1][1] = zc;

  // Prologue: stage slab 0.
  load_tile(0);
  store_tile(0);
  __syncthreads();

  for (int kk = 0; kk < 1024; kk += 32) {
    const int  cur  = (kk >> 5) & 1;
    const bool more = (kk + 32) < 1024;

    if (more) load_tile(kk + 32);   // global loads in flight during WMMA

    v16bf a0 = load_frag(&As[cur][(wm0 + 0)  * 32], 32);
    v16bf a1 = load_frag(&As[cur][(wm0 + 16) * 32], 32);
    v16bf b0 = load_frag(&Bs[cur][(wn0 + 0)  * 32], 32);
    v16bf b1 = load_frag(&Bs[cur][(wn0 + 16) * 32], 32);
    acc[0][0] = bf_wmma(a0, b0, acc[0][0]);
    acc[0][1] = bf_wmma(a0, b1, acc[0][1]);
    acc[1][0] = bf_wmma(a1, b0, acc[1][0]);
    acc[1][1] = bf_wmma(a1, b1, acc[1][1]);

    if (more) {
      store_tile(cur ^ 1);          // other buffer: no hazard with readers
      __syncthreads();              // single barrier per k-step
    }
  }

  // Epilogue: bias add, bf16 convert, scatter to attention-friendly layouts.
  const int hfl = lane >> 4;
  const int nl  = lane & 15;
  #pragma unroll
  for (int nt = 0; nt < 2; ++nt) {
    const int ng = nbase + wn0 + nt * 16 + nl;       // global column
    const float bv = bias[ng];
    const int hh = ng >> 6;                          // head
    const int d  = ng & 63;                          // dim within head
    #pragma unroll
    for (int mt = 0; mt < 2; ++mt) {
      #pragma unroll
      for (int i = 0; i < 8; ++i) {
        const int mg = mbase + wm0 + mt * 16 + i + hfl * 8;  // global row
        const int bb = mg >> 11;                              // batch
        const int s  = mg & 2047;                             // seq pos
        const unsigned short bf = f2bf(acc[mt][nt][i] + bv);
        if (proj == 2)
          out[((size_t)(bb * 16 + hh) * 64 + d) * 2048 + s] = bf;   // V^T
        else
          out[((size_t)(bb * 16 + hh) * 2048 + s) * 64 + d] = bf;   // Q, K
      }
    }
  }
}

// ---------------------------------------------------------------------------
// Stage 2: flash attention. One 16-row q tile per wave; 8 waves per block.
// kv processed in 128-wide chunks: S(16x128) = Q(16x64) K^T, online softmax,
// O(16x64) += P(16x128) V. Mask tiles double-buffered via async DMA to LDS.
// ---------------------------------------------------------------------------
__global__ __launch_bounds__(256) void attn_kernel(
    const unsigned short* __restrict__ qws, const unsigned short* __restrict__ kws,
    const unsigned short* __restrict__ vtws, const unsigned char* __restrict__ mask,
    float* __restrict__ outp)
{
  __shared__ unsigned short p_lds[8][16 * 128];     // per-wave P staging (32 KB)
  __shared__ unsigned char  m_lds[8][2][16 * 128];  // per-wave mask ping-pong (32 KB)

  const int tid  = threadIdx.x;
  const int wave = tid >> 5;
  const int lane = tid & 31;
  const int g  = blockIdx.x * 8 + wave;   // global wave id: 8192 q-tiles
  const int bh = g >> 7;                  // 128 q-tiles per (b,h)
  const int qt = g & 127;
  const int b  = bh >> 4;
  const int h  = bh & 15;
  const int q0 = qt * 16;

  const unsigned short* qbase = qws  + ((size_t)bh * 2048 + q0) * 64;
  const unsigned short* kbase = kws  + (size_t)bh * 2048 * 64;
  const unsigned short* vbase = vtws + (size_t)bh * 64 * 2048;
  const unsigned char*  msk   = mask + ((size_t)b * 2048 + q0) * 2048;

  const int hfl = lane >> 4;
  const int nl  = lane & 15;
  unsigned short* pl = &p_lds[wave][0];
  // LDS byte offset of this wave's mask buffers (generic ptr low 32 bits =
  // LDS offset per the aperture mapping, ISA 10.2).
  const unsigned mlds_off = (unsigned)(uintptr_t)&m_lds[wave][0][0];
  const int mrow4  = lane >> 3;           // async-copy lane mapping: 4 rows /
  const int mcol16 = (lane & 7) * 16;     // instr, 8 lanes x 16B per row

  // Issue the 4 async b128 loads that move one 16x128 mask tile into LDS.
  auto issue_mask_dma = [&](int kvc, int buf) {
    #pragma unroll
    for (int r4 = 0; r4 < 4; ++r4) {
      const int row = r4 * 4 + mrow4;
      const unsigned long long ga =
          (unsigned long long)(uintptr_t)(msk + (size_t)row * 2048 + kvc + mcol16);
      const unsigned la = mlds_off + (unsigned)(buf * 2048 + row * 128 + mcol16);
      asm volatile("global_load_async_to_lds_b128 %0, %1, off"
                   :: "v"(la), "v"(ga) : "memory");
    }
  };

  // Q fragments for d = 0..31 and 32..63 (reused across the whole kv loop)
  const v16bf qf0 = load_frag(qbase,      64);
  const v16bf qf1 = load_frag(qbase + 32, 64);

  const v8f zc = {};
  float mrow[8], lrow[8];
  v8f oacc[4];
  #pragma unroll
  for (int i = 0; i < 8; ++i) { mrow[i] = -3.0e38f; lrow[i] = 0.0f; }
  #pragma unroll
  for (int dt = 0; dt < 4; ++dt) oacc[dt] = zc;

  issue_mask_dma(0, 0);                   // prologue: chunk 0 mask in flight

  for (int kv0 = 0; kv0 < 2048; kv0 += 128) {
    const int cbuf = (kv0 >> 7) & 1;

    // ---- S = Q K^T over this 128-wide chunk (8 n-tiles x 2 k-chunks) ----
    v8f sacc[8];
    #pragma unroll
    for (int nt = 0; nt < 8; ++nt) {
      const unsigned short* kb = kbase + (size_t)(kv0 + nt * 16) * 64;
      v8f c = zc;
      c = bf_wmma(qf0, load_frag(kb,      64), c);
      c = bf_wmma(qf1, load_frag(kb + 32, 64), c);
      sacc[nt] = c;
    }

    // ---- current mask tile ready; immediately start next chunk's DMA ----
    asm volatile("s_wait_asynccnt 0x0" ::: "memory");
    issue_mask_dma((kv0 + 128) & 2047, cbuf ^ 1);   // branchless wrap; the
                                                    // final extra issue is
                                                    // absorbed by S_ENDPGM's
                                                    // implicit wait-idle

    // ---- scale + mask from LDS ----
    const unsigned char* ml = &m_lds[wave][cbuf][0];
    #pragma unroll
    for (int nt = 0; nt < 8; ++nt) {
      const int n = nt * 16 + nl;
      #pragma unroll
      for (int i = 0; i < 8; ++i) {
        const int m = i + hfl * 8;
        const unsigned char mk = ml[m * 128 + n];
        sacc[nt][i] = mk ? NEGV : sacc[nt][i] * 0.125f;
      }
    }

    // ---- online softmax: row max / exp / row sum (16-lane butterflies) ----
    float corr[8];
    #pragma unroll
    for (int i = 0; i < 8; ++i) {
      float t = sacc[0][i];
      #pragma unroll
      for (int nt = 1; nt < 8; ++nt) t = fmaxf(t, sacc[nt][i]);
      #pragma unroll
      for (int off = 1; off < 16; off <<= 1)
        t = fmaxf(t, __shfl_xor(t, off, 32));
      const float mnew = fmaxf(mrow[i], t);
      corr[i] = __expf(mrow[i] - mnew);
      mrow[i] = mnew;
    }
    #pragma unroll
    for (int i = 0; i < 8; ++i) {
      float rs = 0.0f;
      #pragma unroll
      for (int nt = 0; nt < 8; ++nt) {
        const float p = __expf(sacc[nt][i] - mrow[i]);
        sacc[nt][i] = p;
        rs += p;
      }
      #pragma unroll
      for (int off = 1; off < 16; off <<= 1)
        rs += __shfl_xor(rs, off, 32);
      lrow[i] = lrow[i] * corr[i] + rs;
      #pragma unroll
      for (int dt = 0; dt < 4; ++dt) oacc[dt][i] *= corr[i];
    }

    // ---- P: C-layout regs -> LDS row-major [m][kv_local] as bf16 ----
    // (LDS ops are in-order within a wave; region is wave-private.)
    #pragma unroll
    for (int nt = 0; nt < 8; ++nt) {
      const int col = nt * 16 + nl;
      #pragma unroll
      for (int i = 0; i < 8; ++i)
        pl[(i + hfl * 8) * 128 + col] = f2bf(sacc[nt][i]);
    }

    // ---- O += P V (4 k-chunks x 4 d-tiles) ----
    #pragma unroll
    for (int kc = 0; kc < 4; ++kc) {
      const v16bf pf = load_frag(pl + kc * 32, 128);
      #pragma unroll
      for (int dt = 0; dt < 4; ++dt) {
        const unsigned short* vb =
            vbase + (size_t)(dt * 16) * 2048 + kv0 + kc * 32;
        oacc[dt] = bf_wmma(pf, load_frag(vb, 2048), oacc[dt]);
      }
    }
  }

  // ---- epilogue: normalize and scatter to [b, sq, h*64 + d] fp32 ----
  #pragma unroll
  for (int i = 0; i < 8; ++i) {
    const float inv = 1.0f / lrow[i];
    const int m = q0 + i + hfl * 8;
    float* orow = outp + ((size_t)b * 2048 + m) * 1024 + h * 64;
    #pragma unroll
    for (int dt = 0; dt < 4; ++dt)
      orow[dt * 16 + nl] = oacc[dt][i] * inv;
  }
}

extern "C" void kernel_launch(void* const* d_in, const int* in_sizes, int n_in,
                              void* d_out, int out_size, void* d_ws, size_t ws_size,
                              hipStream_t stream) {
  (void)in_sizes; (void)n_in; (void)out_size; (void)ws_size;
  const float* x_q  = (const float*)d_in[0];
  const float* x_kv = (const float*)d_in[1];
  const unsigned char* mask = (const unsigned char*)d_in[2];  // jnp.bool_ = 1B
  const float* w_q = (const float*)d_in[3];
  const float* b_q = (const float*)d_in[4];
  const float* w_k = (const float*)d_in[5];
  const float* b_k = (const float*)d_in[6];
  const float* w_v = (const float*)d_in[7];
  const float* b_v = (const float*)d_in[8];
  float* out = (float*)d_out;

  // Workspace: bf16 Q [b,h,s,64], K [b,h,s,64], V^T [b,h,64,s] (16.78 MB each)
  const size_t elems = (size_t)4 * 16 * 2048 * 64;   // 8,388,608
  unsigned short* qws  = (unsigned short*)d_ws;
  unsigned short* kws  = qws + elems;
  unsigned short* vtws = kws + elems;

  // Stage 1: 3 projections, tiles 128x64 over 8192x1024.
  dim3 pgrid(1024 / 64, 8192 / 128, 3);
  qkv_proj_kernel<<<pgrid, 256, 0, stream>>>(x_q, x_kv, w_q, w_k, w_v,
                                             b_q, b_k, b_v, qws, kws, vtws);

  // Stage 2: 8192 q-tiles, 8 waves per block.
  attn_kernel<<<dim3(1024), 256, 0, stream>>>(qws, kws, vtws, mask, out);
}
